// Layer_74311524155837
// MI455X (gfx1250) — matrix operand order repaired and checked by reading, not compiled, not run on hardware
//
#include <hip/hip_runtime.h>
#include <math.h>
#include <stdint.h>

// ---------------------------------------------------------------------------
// Types for CDNA5 WMMA (wave32, v_wmma_f32_16x16x32_bf16)
// ---------------------------------------------------------------------------
typedef __bf16 bf16_t;
typedef __attribute__((ext_vector_type(8)))  __bf16 v8bf;
typedef __attribute__((ext_vector_type(16))) __bf16 v16bf;
typedef __attribute__((ext_vector_type(8)))  float  v8f;

static __device__ __forceinline__ v16bf cat8(v8bf lo, v8bf hi) {
  return __builtin_shufflevector(lo, hi, 0,1,2,3,4,5,6,7,8,9,10,11,12,13,14,15);
}

static __device__ __forceinline__ v8f wmma_bf16(v16bf a, v16bf b, v8f c) {
  // (neg_a, A, neg_b, B, c_mod, C, reuse_a, reuse_b)
  return __builtin_amdgcn_wmma_f32_16x16x32_bf16(false, a, false, b, (short)0, c,
                                                 false, false);
}

// LDS byte offset of a __shared__ object: generic LDS addresses are
// {aperture_hi32, lds_offset32}, so truncation yields the DS offset.
static __device__ __forceinline__ uint32_t lds_off_u32(const void* p) {
  return (uint32_t)(uintptr_t)p;
}

// GLOBAL_LOAD_ASYNC_TO_LDS_B128, GVS mode: mem = SGPR64 + VGPR32 + 0,
// LDS[vdst + 0..15] = MEM[...].  Tracked by ASYNCcnt.
static __device__ __forceinline__ void async_ld_b128(uint32_t lds_off,
                                                     uint32_t gbyte_off,
                                                     const void* gbase) {
  asm volatile("global_load_async_to_lds_b128 %0, %1, %2"
               :
               : "v"(lds_off), "v"(gbyte_off),
                 "s"((unsigned long long)(uintptr_t)gbase)
               : "memory");
}

static __device__ __forceinline__ void wait_async0() {
  asm volatile("s_wait_asynccnt 0x0" ::: "memory");
}

// ---------------------------------------------------------------------------
// Problem constants
// ---------------------------------------------------------------------------
#define BSZ   32
#define TSEQ  512
#define EMB   384
#define HEADS 6
#define QD    64
#define DFF   1536
#define NTOK  (BSZ * TSEQ)   // 16384

// workspace layout (bytes), all 256B aligned
#define OFF_WQKV 0UL                                  // 384*1152 bf16
#define OFF_WP   (OFF_WQKV + 884736UL)                // 384*384  bf16
#define OFF_W1   (OFF_WP   + 294912UL)                // 384*1536 bf16
#define OFF_W2   (OFF_W1   + 1179648UL)               // 1536*384 bf16
#define OFF_H    (OFF_W2   + 1179648UL)               // 16384*384 bf16
#define OFF_Q    (OFF_H    + 12582912UL)              // [B,H,T,D] bf16
#define OFF_K    (OFF_Q    + 12582912UL)
#define OFF_V    (OFF_K    + 12582912UL)
#define OFF_O    (OFF_V    + 12582912UL)              // 16384*384 bf16
#define OFF_X1   (OFF_O    + 12582912UL)              // 16384*384 f32
#define OFF_H2   (OFF_X1   + 25165824UL)              // 16384*384 bf16
#define OFF_FF1  (OFF_H2   + 12582912UL)              // 16384*1536 bf16

// ---------------------------------------------------------------------------
// Weight conversion: f32 -> bf16; Wq|Wk|Wv fused to [384 x 1152]
// ---------------------------------------------------------------------------
__global__ void cvt_weights_kernel(const float* __restrict__ Wq,
                                   const float* __restrict__ Wk,
                                   const float* __restrict__ Wv,
                                   const float* __restrict__ Wp,
                                   const float* __restrict__ W1,
                                   const float* __restrict__ W2,
                                   bf16_t* __restrict__ wqkv,
                                   bf16_t* __restrict__ wp,
                                   bf16_t* __restrict__ w1,
                                   bf16_t* __restrict__ w2) {
  int i = blockIdx.x * blockDim.x + threadIdx.x;
  if (i < EMB * EMB) {
    int c = i / EMB, j = i - c * EMB;
    wqkv[c * (3 * EMB) + j]           = (bf16_t)Wq[i];
    wqkv[c * (3 * EMB) + EMB + j]     = (bf16_t)Wk[i];
    wqkv[c * (3 * EMB) + 2 * EMB + j] = (bf16_t)Wv[i];
    wp[i] = (bf16_t)Wp[i];
  }
  if (i < EMB * DFF) {
    w1[i] = (bf16_t)W1[i];
    w2[i] = (bf16_t)W2[i];
  }
}

// ---------------------------------------------------------------------------
// LayerNorm over 384-wide rows, output bf16.  One block (128 threads) per row.
// ---------------------------------------------------------------------------
__global__ void __launch_bounds__(128)
ln_bf16_kernel(const float* __restrict__ x, const float* __restrict__ gam,
               const float* __restrict__ bet, bf16_t* __restrict__ out) {
  __shared__ float red[4];
  const int row = blockIdx.x;
  const int tid = threadIdx.x;
  const float* xr = x + (size_t)row * EMB;
  float v0 = xr[tid], v1 = xr[tid + 128], v2 = xr[tid + 256];
  float s = v0 + v1 + v2;
  #pragma unroll
  for (int m = 16; m >= 1; m >>= 1) s += __shfl_xor(s, m, 32);
  if ((tid & 31) == 0) red[tid >> 5] = s;
  __syncthreads();
  float mu = (red[0] + red[1] + red[2] + red[3]) * (1.0f / EMB);
  float d0 = v0 - mu, d1 = v1 - mu, d2 = v2 - mu;
  float s2 = d0 * d0 + d1 * d1 + d2 * d2;
  __syncthreads();
  #pragma unroll
  for (int m = 16; m >= 1; m >>= 1) s2 += __shfl_xor(s2, m, 32);
  if ((tid & 31) == 0) red[tid >> 5] = s2;
  __syncthreads();
  float var = (red[0] + red[1] + red[2] + red[3]) * (1.0f / EMB);
  float rr = rsqrtf(var + 1e-5f);
  bf16_t* orow = out + (size_t)row * EMB;
  orow[tid]       = (bf16_t)(d0 * rr * gam[tid]       + bet[tid]);
  orow[tid + 128] = (bf16_t)(d1 * rr * gam[tid + 128] + bet[tid + 128]);
  orow[tid + 256] = (bf16_t)(d2 * rr * gam[tid + 256] + bet[tid + 256]);
}

// ---------------------------------------------------------------------------
// WMMA GEMM:  C[M x N] = A[M x K] * B[K x N]  (bf16 in, f32 acc)
//   block tile 128x64, 256 threads = 8 waves arranged 4(M) x 2(N),
//   each wave owns a 32x32 tile = 2x2 v_wmma_f32_16x16x32_bf16 accumulators.
//   Double-buffered LDS; A tiles staged with global_load_async_to_lds_b128
//   (ASYNCcnt), B tiles staged via VGPR + ds scatter (transposed for
//   contiguous-K fragments).  Tile k+1 copy overlaps tile k WMMAs.
// Epilogue MODE: 0=QKV scatter(bf16), 1=proj+bias+resid(f32),
//                2=ffn1+bias+relu(bf16), 3=ffn2+bias+resid(f32)
// ---------------------------------------------------------------------------
template <int K, int N, int MODE>
__global__ void __launch_bounds__(256)
gemm_wmma_kernel(const bf16_t* __restrict__ A, const bf16_t* __restrict__ Bm,
                 const float* __restrict__ bias, const float* __restrict__ resid,
                 float* __restrict__ outf,
                 bf16_t* __restrict__ ob0, bf16_t* __restrict__ ob1,
                 bf16_t* __restrict__ ob2) {
  __shared__ bf16_t As[2][128 * 40];   // [row][k], pad->80B rows (16B aligned)
  __shared__ bf16_t BsT[2][64 * 40];   // [col][k] transposed
  const int tid  = threadIdx.x;
  const int lane = tid & 31, wave = tid >> 5;
  const int wm = wave >> 1, wn = wave & 1;
  const int g = lane >> 4, nn = lane & 15;
  const int rowBase = blockIdx.x * 128;
  const int colBase = blockIdx.y * 64;
  v8f acc[2][2] = {};
  const int ar = tid >> 2, ac = (tid & 3) * 8;   // A loader: 2 x 16B / thread
  const int bk = tid >> 3, bc = (tid & 7) * 8;   // B loader: 1 x 16B / thread

  const size_t aRow0 = (size_t)(rowBase + ar) * K + ac;        // elements
  const size_t aRow1 = (size_t)(rowBase + 64 + ar) * K + ac;
  const uint32_t ldsA0[2] = {lds_off_u32(&As[0][ar * 40 + ac]),
                             lds_off_u32(&As[1][ar * 40 + ac])};
  const uint32_t ldsA1[2] = {lds_off_u32(&As[0][(64 + ar) * 40 + ac]),
                             lds_off_u32(&As[1][(64 + ar) * 40 + ac])};

  const int NK = K / 32;
  // ---- prologue: stage tile 0 into buffer 0 ----
  async_ld_b128(ldsA0[0], (uint32_t)((aRow0 + 0) * 2), A);
  async_ld_b128(ldsA1[0], (uint32_t)((aRow1 + 0) * 2), A);
  v8bf bv = *(const v8bf*)(Bm + (size_t)bk * N + colBase + bc);
  #pragma unroll
  for (int j = 0; j < 8; ++j) BsT[0][(bc + j) * 40 + bk] = bv[j];
  wait_async0();
  __syncthreads();

  for (int kt = 0; kt < NK; ++kt) {
    const int cur = kt & 1, nxt = cur ^ 1;
    const bool more = (kt + 1 < NK);
    if (more) {
      // prefetch tile kt+1 while computing tile kt
      const int k1 = (kt + 1) * 32;
      async_ld_b128(ldsA0[nxt], (uint32_t)((aRow0 + k1) * 2), A);
      async_ld_b128(ldsA1[nxt], (uint32_t)((aRow1 + k1) * 2), A);
      bv = *(const v8bf*)(Bm + (size_t)(k1 + bk) * N + colBase + bc);
      if (kt + 2 < NK)
        __builtin_prefetch(Bm + (size_t)((kt + 2) * 32 + bk) * N + colBase + bc,
                           0, 0);
    }

    v16bf afrag[2], bfrag[2];
    #pragma unroll
    for (int t = 0; t < 2; ++t) {
      // A fragment: lane holds row (lane&15); K(e)=8g+e (e<8), 8g+e+8 (e>=8)
      int aoff = (wm * 32 + t * 16 + nn) * 40 + 8 * g;
      afrag[t] = cat8(*(const v8bf*)&As[cur][aoff],
                      *(const v8bf*)&As[cur][aoff + 16]);
      // B fragment: lane holds col (lane&15); K(e)=8g+e (e<8), 8g+e+8 (e>=8)
      int boff = (wn * 32 + t * 16 + nn) * 40 + 8 * g;
      bfrag[t] = cat8(*(const v8bf*)&BsT[cur][boff],
                      *(const v8bf*)&BsT[cur][boff + 16]);
    }
    #pragma unroll
    for (int ti = 0; ti < 2; ++ti)
      #pragma unroll
      for (int tj = 0; tj < 2; ++tj)
        acc[ti][tj] = wmma_bf16(afrag[ti], bfrag[tj], acc[ti][tj]);

    if (more) {
      #pragma unroll
      for (int j = 0; j < 8; ++j) BsT[nxt][(bc + j) * 40 + bk] = bv[j];
    }
    wait_async0();
    __syncthreads();
  }

  // Epilogue.  C/D layout: element r of lane (g,nn) = (m = r+8g, n = nn).
  #pragma unroll
  for (int ti = 0; ti < 2; ++ti) {
    #pragma unroll
    for (int tj = 0; tj < 2; ++tj) {
      #pragma unroll
      for (int r = 0; r < 8; ++r) {
        int row = rowBase + wm * 32 + ti * 16 + r + 8 * g;
        int col = colBase + wn * 32 + tj * 16 + nn;
        float val = acc[ti][tj][r];
        if (MODE == 0) {
          int which = col / EMB;
          int rem = col - which * EMB;
          int hh = rem >> 6, dd = rem & 63;
          int bb = row >> 9, tt = row & 511;
          size_t dst = (((size_t)bb * HEADS + hh) * TSEQ + tt) * QD + dd;
          bf16_t wv = (bf16_t)val;
          if (which == 0)      ob0[dst] = wv;
          else if (which == 1) ob1[dst] = wv;
          else                 ob2[dst] = wv;
        } else if (MODE == 1 || MODE == 3) {
          outf[(size_t)row * N + col] =
              resid[(size_t)row * N + col] + val + bias[col];
        } else {  // MODE == 2
          float u = val + bias[col];
          ob0[(size_t)row * N + col] = (bf16_t)(u > 0.f ? u : 0.f);
        }
      }
    }
  }
}

// ---------------------------------------------------------------------------
// Flash attention (causal).  One block = one (b,h) x 128 query rows.
// 8 waves, each owns 16 q-rows and the full D=64 output (4 f32 accumulators).
// Q and K tiles staged with async-to-LDS copies; V via VGPR scatter-transpose.
// Per 32-key step: S = Q K^T (4 WMMAs), online softmax via 16-lane shuffles,
// P relayout through LDS, O += P V (4 WMMAs).
// ---------------------------------------------------------------------------
__global__ void __launch_bounds__(256)
attn_kernel(const bf16_t* __restrict__ q, const bf16_t* __restrict__ k,
            const bf16_t* __restrict__ v, bf16_t* __restrict__ o) {
  __shared__ bf16_t Qs[128 * 72];     // [qrow][d]
  __shared__ bf16_t Ks[32 * 72];      // [key][d]
  __shared__ bf16_t VsT[64 * 40];     // [d][key]
  __shared__ bf16_t Ps[8 * 16 * 40];  // per-wave P tile [m][key]
  const int tid  = threadIdx.x;
  const int lane = tid & 31, wave = tid >> 5;
  const int g = lane >> 4, nn = lane & 15;
  const int qb = blockIdx.x & 3;
  const int bh = blockIdx.x >> 2;
  const size_t bhbase = (size_t)bh * TSEQ;
  const int rowbase = qb * 128 + wave * 16;

  // stage Q block [128 x 64] via async copies (4 x 16B per thread)
  #pragma unroll
  for (int p = 0; p < 4; ++p) {
    int idx = (p * 256 + tid) * 8;
    int rr = idx >> 6, cc = idx & 63;
    async_ld_b128(lds_off_u32(&Qs[rr * 72 + cc]),
                  (uint32_t)(((bhbase + qb * 128 + rr) * QD + cc) * 2), q);
  }

  v8f acc[4] = {};
  float m_i[8], l_i[8];
  #pragma unroll
  for (int r = 0; r < 8; ++r) { m_i[r] = -1e30f; l_i[r] = 0.f; }

  const int kr = tid >> 3, kc = (tid & 7) * 8;
  const uint32_t ldsK = lds_off_u32(&Ks[kr * 72 + kc]);
  const int nkb = (qb + 1) * 4;
  for (int kb = 0; kb < nkb; ++kb) {
    __syncthreads();  // previous K/V fully consumed
    async_ld_b128(ldsK, (uint32_t)(((bhbase + kb * 32 + kr) * QD + kc) * 2), k);
    v8bf vv = *(const v8bf*)(v + (bhbase + kb * 32 + kr) * QD + kc);
    #pragma unroll
    for (int j = 0; j < 8; ++j) VsT[(kc + j) * 40 + kr] = vv[j];
    wait_async0();    // covers Q staging on first iteration too
    __syncthreads();

    // scores S = Q K^T over D=64 (two 32-wide K steps)
    v8f s0 = {}, s1 = {};
    #pragma unroll
    for (int ds = 0; ds < 2; ++ds) {
      int aoff = (wave * 16 + nn) * 72 + ds * 32 + 8 * g;
      v16bf aq = cat8(*(const v8bf*)&Qs[aoff], *(const v8bf*)&Qs[aoff + 16]);
      int b0 = nn * 72 + ds * 32 + 8 * g;
      v16bf bk0 = cat8(*(const v8bf*)&Ks[b0], *(const v8bf*)&Ks[b0 + 16]);
      int b1 = (16 + nn) * 72 + ds * 32 + 8 * g;
      v16bf bk1 = cat8(*(const v8bf*)&Ks[b1], *(const v8bf*)&Ks[b1 + 16]);
      s0 = wmma_bf16(aq, bk0, s0);
      s1 = wmma_bf16(aq, bk1, s1);
    }

    // online softmax: rows live across 16 lanes of each half-wave
    #pragma unroll
    for (int r = 0; r < 8; ++r) {
      int qrow = rowbase + r + 8 * g;
      int key0 = kb * 32 + nn;
      float e0 = s0[r] * 0.125f;   // 1/sqrt(64)
      float e1 = s1[r] * 0.125f;
      if (key0 > qrow)      e0 = -1e30f;
      if (key0 + 16 > qrow) e1 = -1e30f;
      float mx = fmaxf(e0, e1);
      #pragma unroll
      for (int mo = 8; mo >= 1; mo >>= 1)
        mx = fmaxf(mx, __shfl_xor(mx, mo, 16));
      float mnew = fmaxf(m_i[r], mx);
      float sc = __expf(m_i[r] - mnew);
      float p0 = __expf(e0 - mnew);
      float p1 = __expf(e1 - mnew);
      float rs = p0 + p1;
      #pragma unroll
      for (int mo = 8; mo >= 1; mo >>= 1)
        rs += __shfl_xor(rs, mo, 16);
      l_i[r] = l_i[r] * sc + rs;
      m_i[r] = mnew;
      #pragma unroll
      for (int j = 0; j < 4; ++j) acc[j][r] *= sc;
      Ps[(wave * 16 + r + 8 * g) * 40 + nn]      = (bf16_t)p0;
      Ps[(wave * 16 + r + 8 * g) * 40 + 16 + nn] = (bf16_t)p1;
    }

    // O += P V  (wave-local LDS round trip relayouts P into A-fragment form)
    int poff = (wave * 16 + nn) * 40 + 8 * g;
    v16bf pf = cat8(*(const v8bf*)&Ps[poff], *(const v8bf*)&Ps[poff + 16]);
    #pragma unroll
    for (int j = 0; j < 4; ++j) {
      int voff = (j * 16 + nn) * 40 + 8 * g;
      v16bf vf = cat8(*(const v8bf*)&VsT[voff], *(const v8bf*)&VsT[voff + 16]);
      acc[j] = wmma_bf16(pf, vf, acc[j]);
    }
  }

  // normalize and scatter to token-major [NTOK x EMB] bf16
  int b = bh / HEADS, h = bh - b * HEADS;
  #pragma unroll
  for (int r = 0; r < 8; ++r) {
    float inv = 1.0f / l_i[r];
    int token = b * TSEQ + rowbase + r + 8 * g;
    #pragma unroll
    for (int j = 0; j < 4; ++j)
      o[(size_t)token * EMB + h * QD + j * 16 + nn] = (bf16_t)(acc[j][r] * inv);
  }
}

// ---------------------------------------------------------------------------
// Host-side launch
// ---------------------------------------------------------------------------
extern "C" void kernel_launch(void* const* d_in, const int* in_sizes, int n_in,
                              void* d_out, int out_size, void* d_ws,
                              size_t ws_size, hipStream_t stream) {
  (void)in_sizes; (void)n_in; (void)out_size; (void)ws_size;
  const float* x   = (const float*)d_in[0];
  const float* Wq  = (const float*)d_in[1];
  const float* Wk  = (const float*)d_in[2];
  const float* Wv  = (const float*)d_in[3];
  const float* Wp  = (const float*)d_in[4];
  const float* bp  = (const float*)d_in[5];
  const float* g1  = (const float*)d_in[6];
  const float* b1  = (const float*)d_in[7];
  const float* g2  = (const float*)d_in[8];
  const float* b2  = (const float*)d_in[9];
  const float* W1  = (const float*)d_in[10];
  const float* bf1 = (const float*)d_in[11];
  const float* W2  = (const float*)d_in[12];
  const float* bf2 = (const float*)d_in[13];
  float* out = (float*)d_out;

  char* w = (char*)d_ws;
  bf16_t* wqkv = (bf16_t*)(w + OFF_WQKV);
  bf16_t* wp   = (bf16_t*)(w + OFF_WP);
  bf16_t* w1   = (bf16_t*)(w + OFF_W1);
  bf16_t* w2   = (bf16_t*)(w + OFF_W2);
  bf16_t* hbuf = (bf16_t*)(w + OFF_H);
  bf16_t* qbuf = (bf16_t*)(w + OFF_Q);
  bf16_t* kbuf = (bf16_t*)(w + OFF_K);
  bf16_t* vbuf = (bf16_t*)(w + OFF_V);
  bf16_t* obuf = (bf16_t*)(w + OFF_O);
  float*  x1   = (float*) (w + OFF_X1);
  bf16_t* h2   = (bf16_t*)(w + OFF_H2);
  bf16_t* ff1  = (bf16_t*)(w + OFF_FF1);

  // 1) weights -> bf16 (Wq|Wk|Wv fused)
  cvt_weights_kernel<<<(EMB * DFF + 255) / 256, 256, 0, stream>>>(
      Wq, Wk, Wv, Wp, W1, W2, wqkv, wp, w1, w2);
  // 2) LN1
  ln_bf16_kernel<<<NTOK, 128, 0, stream>>>(x, g1, b1, hbuf);
  // 3) fused QKV projection: [16384x384] x [384x1152]
  gemm_wmma_kernel<EMB, 3 * EMB, 0>
      <<<dim3(NTOK / 128, (3 * EMB) / 64), 256, 0, stream>>>(
          hbuf, wqkv, nullptr, nullptr, nullptr, qbuf, kbuf, vbuf);
  // 4) causal flash attention
  attn_kernel<<<BSZ * HEADS * (TSEQ / 128), 256, 0, stream>>>(qbuf, kbuf, vbuf,
                                                              obuf);
  // 5) out projection + bias + residual -> x1 (f32)
  gemm_wmma_kernel<EMB, EMB, 1>
      <<<dim3(NTOK / 128, EMB / 64), 256, 0, stream>>>(
          obuf, wp, bp, x, x1, nullptr, nullptr, nullptr);
  // 6) LN2
  ln_bf16_kernel<<<NTOK, 128, 0, stream>>>(x1, g2, b2, h2);
  // 7) FFN1 + bias + ReLU -> bf16
  gemm_wmma_kernel<EMB, DFF, 2>
      <<<dim3(NTOK / 128, DFF / 64), 256, 0, stream>>>(
          h2, w1, bf1, nullptr, nullptr, ff1, nullptr, nullptr);
  // 8) FFN2 + bias + residual -> d_out (f32)
  gemm_wmma_kernel<DFF, EMB, 3>
      <<<dim3(NTOK / 128, EMB / 64), 256, 0, stream>>>(
          ff1, w2, bf2, x1, out, nullptr, nullptr, nullptr);
}
